// QNetwork_67800353735333
// MI455X (gfx1250) — matrix operand order repaired and checked by reading, not compiled, not run on hardware
//
#include <hip/hip_runtime.h>

// ---------------------------------------------------------------------------
// Problem constants (from reference): B=512, T=256, OBS=512, H=128, A=6
// ---------------------------------------------------------------------------
#define BB   512
#define TT   256
#define OBS  512
#define HH   128
#define AA   6
#define BT   (BB * TT)          // 131072 rows
#define H4   (4 * HH)           // 512

typedef __attribute__((ext_vector_type(16))) __bf16 v16bf;
typedef __attribute__((ext_vector_type(8)))  float  v8f;
typedef __attribute__((ext_vector_type(4)))  unsigned int uintx4;
typedef __attribute__((ext_vector_type(4)))  int  intx4;
typedef __attribute__((ext_vector_type(8)))  int  intx8;

// ---------------------------------------------------------------------------
// bf16 helpers (kept as unsigned short everywhere except the WMMA boundary)
// ---------------------------------------------------------------------------
static __device__ __forceinline__ unsigned short f2bf(float f) {
    unsigned u = __float_as_uint(f);
    u += 0x7FFFu + ((u >> 16) & 1u);          // round-to-nearest-even
    return (unsigned short)(u >> 16);
}
static __device__ __forceinline__ float bf2f(unsigned short s) {
    return __uint_as_float(((unsigned)s) << 16);
}

union FragU {
    uint4          q[2];
    unsigned short s[16];
    v16bf          v;
};

// A-frag (16x32, row major, k contiguous):  lane holds M=lane&15,
//   k = kbase + 8*half + {0..7}  and  kbase + 16 + 8*half + {0..7}
static __device__ __forceinline__ v16bf load_a_bf16(const unsigned short* rowp,
                                                    int kbase, int half) {
    FragU u;
    u.q[0] = *(const uint4*)(rowp + kbase + 8 * half);
    u.q[1] = *(const uint4*)(rowp + kbase + 16 + 8 * half);
    return u.v;
}
// B-frag (32x16) from weights stored as [N][K] bf16 (k contiguous):
//   lane holds N=lane&15, k = kbase + 8*half + {0..7}  and  kbase + 16 + 8*half + {0..7}
static __device__ __forceinline__ v16bf load_b_bf16(const unsigned short* colp,
                                                    int kbase, int half) {
    FragU u;
    u.q[0] = *(const uint4*)(colp + kbase + 8 * half);
    u.q[1] = *(const uint4*)(colp + kbase + 16 + 8 * half);
    return u.v;
}
// A-frag built from f32 source (x), converting to bf16 on the fly.
static __device__ __forceinline__ v16bf load_a_f32(const float* rowp,
                                                   int kbase, int half) {
    const float4 f0 = *(const float4*)(rowp + kbase + 8 * half);
    const float4 f1 = *(const float4*)(rowp + kbase + 8 * half + 4);
    const float4 f2 = *(const float4*)(rowp + kbase + 16 + 8 * half);
    const float4 f3 = *(const float4*)(rowp + kbase + 16 + 8 * half + 4);
    FragU u;
    u.s[0] = f2bf(f0.x);  u.s[1] = f2bf(f0.y);  u.s[2] = f2bf(f0.z);  u.s[3] = f2bf(f0.w);
    u.s[4] = f2bf(f1.x);  u.s[5] = f2bf(f1.y);  u.s[6] = f2bf(f1.z);  u.s[7] = f2bf(f1.w);
    u.s[8] = f2bf(f2.x);  u.s[9] = f2bf(f2.y);  u.s[10]= f2bf(f2.z);  u.s[11]= f2bf(f2.w);
    u.s[12]= f2bf(f3.x);  u.s[13]= f2bf(f3.y);  u.s[14]= f2bf(f3.z);  u.s[15]= f2bf(f3.w);
    return u.v;
}

static __device__ __forceinline__ v8f wmma_bf16(v16bf a, v16bf b, v8f c) {
    return __builtin_amdgcn_wmma_f32_16x16x32_bf16(
        /*neg_a=*/false, a, /*neg_b=*/false, b,
        /*c_mod=*/(short)0, c, /*reuse_a=*/false, /*reuse_b=*/false);
}

static __device__ __forceinline__ float sigm(float x) {
    return 1.0f / (1.0f + __expf(-x));
}
static __device__ __forceinline__ float tanh_fast(float x) {
    float e = __expf(2.0f * x);
    return 1.0f - 2.0f / (e + 1.0f);
}

// ---------------------------------------------------------------------------
// TDM: 2D tensor load into LDS.  D# layout per CDNA5 ISA 08_async_tensor §8.
//   group0: count=1, lds_addr[63:32], global_addr[120:64], type=2[127:126]
//   group1: data_size[17:16], tensor_dim0[79:48], tensor_dim1[111:80],
//           tile_dim0[127:112], tile_dim1[143:128], tensor_dim0_stride[207:160]
// ---------------------------------------------------------------------------
static __device__ __forceinline__ void tdm_load_2d(const void* gptr, unsigned lds_addr,
                                                   unsigned dim0, unsigned dim1,
                                                   unsigned tile0, unsigned tile1,
                                                   unsigned stride0, unsigned dsz_code) {
    unsigned long long ga = (unsigned long long)(size_t)gptr;
    uintx4 g0;
    g0.x = 1u;                                                   // count=1
    g0.y = lds_addr;                                             // LDS byte address
    g0.z = (unsigned)(ga & 0xFFFFFFFFull);                       // global addr lo
    g0.w = (unsigned)((ga >> 32) & 0x01FFFFFFull) | (2u << 30);  // addr hi | type=2
    intx8 g1;
    g1[0] = (int)(dsz_code << 16);                               // data_size
    g1[1] = (int)((dim0 & 0xFFFFu) << 16);                       // tensor_dim0 lo16
    g1[2] = (int)((dim0 >> 16) | ((dim1 & 0xFFFFu) << 16));      // dim0 hi | dim1 lo
    g1[3] = (int)((dim1 >> 16) | (tile0 << 16));                 // dim1 hi | tile_dim0
    g1[4] = (int)(tile1 & 0xFFFFu);                              // tile_dim1 (tile_dim2=0)
    g1[5] = (int)stride0;                                        // dim0 stride lo32
    g1[6] = 0;
    g1[7] = 0;
    intx4 z4 = {0, 0, 0, 0};
#if defined(__clang_major__) && (__clang_major__ >= 23)
    intx8 z8 = {0, 0, 0, 0, 0, 0, 0, 0};
    __builtin_amdgcn_tensor_load_to_lds(g0, g1, z4, z4, z8, 0);
#else
    __builtin_amdgcn_tensor_load_to_lds(g0, g1, z4, z4, 0);
#endif
}

// ---------------------------------------------------------------------------
// Kernel 0: transpose f32 weight [K][N] -> bf16 [Npad][K] (k contiguous)
// ---------------------------------------------------------------------------
__global__ __launch_bounds__(256)
void k_transpose_bf16(const float* __restrict__ src, unsigned short* __restrict__ dst,
                      int K, int N, int Npad) {
    int idx = blockIdx.x * 256 + threadIdx.x;
    if (idx >= Npad * K) return;
    int n = idx / K;
    int k = idx - n * K;
    dst[idx] = (n < N) ? f2bf(src[k * N + n]) : (unsigned short)0;
}

// ---------------------------------------------------------------------------
// Kernel 1: u = relu(x @ W1 + b1) -> bf16 [BT][H]
// One wave per 16-row M tile. K=512 (16 k-frags cached in VGPRs), N=128.
// ---------------------------------------------------------------------------
__global__ __launch_bounds__(256)
void k_gemm1(const float* __restrict__ x, const float* __restrict__ b1,
             const unsigned short* __restrict__ W1t, unsigned short* __restrict__ u_bf) {
    const int tid  = threadIdx.x;
    const int wid  = blockIdx.x * 8 + (tid >> 5);     // wave id = m-tile id
    const int lane = tid & 31;
    const int half = lane >> 4;
    const int ln   = lane & 15;
    const int m0   = wid * 16;

    const float* xrow = x + (size_t)(m0 + ln) * OBS;

    v16bf afr[16];
    #pragma unroll
    for (int kk = 0; kk < 16; ++kk)
        afr[kk] = load_a_f32(xrow, kk * 32, half);

    #pragma unroll
    for (int nt = 0; nt < 8; ++nt) {
        const int n0 = nt * 16;
        const unsigned short* wcol = W1t + (size_t)(n0 + ln) * OBS;
        v8f acc = {};
        #pragma unroll
        for (int kk = 0; kk < 16; ++kk) {
            v16bf bfr = load_b_bf16(wcol, kk * 32, half);
            acc = wmma_bf16(afr[kk], bfr, acc);
        }
        const float bias = b1[n0 + ln];
        #pragma unroll
        for (int r = 0; r < 8; ++r) {
            float v = acc[r] + bias;
            v = fmaxf(v, 0.0f);
            u_bf[(size_t)(m0 + r + 8 * half) * HH + n0 + ln] = f2bf(v);
        }
    }
}

// ---------------------------------------------------------------------------
// Kernel 2: masked LSTM. 32 WGs (one per WGP), each owns 16 batch rows and
// loops T=256 steps.  Wx/Wh (256 KB bf16) are DMAed into LDS once via the
// Tensor Data Mover; per step the z GEMM reads all fragments from LDS.
// Dynamic LDS layout (300 KB total, fits CDNA5's 320 KB/WG):
//   [0      , 131072) wx_s  bf16 [512][128]
//   [131072 , 262144) wh_s  bf16 [512][128]
//   [262144 , 294912) z_s   f32  [16][512]
//   [294912 , 299008) h_s   bf16 [16][128]
//   [299008 , 307200) c_s   f32  [16][128]
// ---------------------------------------------------------------------------
#define LSTM_LDS_BYTES 307200

__global__ __launch_bounds__(256)
void k_lstm(const unsigned short* __restrict__ u_bf,
            const unsigned short* __restrict__ Wxt,
            const unsigned short* __restrict__ Wht,
            const float* __restrict__ bc, const int* __restrict__ mask,
            unsigned short* __restrict__ r_bf) {
    extern __shared__ __align__(16) char smem_raw[];
    unsigned short* wx_s = (unsigned short*)(smem_raw);
    unsigned short* wh_s = (unsigned short*)(smem_raw + 131072);
    float*          z_s  = (float*)         (smem_raw + 262144);
    unsigned short* h_s  = (unsigned short*)(smem_raw + 294912);
    float*          c_s  = (float*)         (smem_raw + 299008);

    const int tid  = threadIdx.x;
    const int wave = tid >> 5;
    const int lane = tid & 31;
    const int half = lane >> 4;
    const int ln   = lane & 15;
    const int b0   = blockIdx.x * 16;

    // elementwise mapping: 16 rows x 16 col-groups of 8
    const int erow = tid >> 4;
    const int j0   = (tid & 15) * 8;

    // ---- one-time: DMA both weight matrices into LDS via the TDM ----
    if (tid == 0) {
        tdm_load_2d(Wxt, (unsigned)(size_t)wx_s, HH, H4, HH, H4, HH, /*2B*/1u);
        tdm_load_2d(Wht, (unsigned)(size_t)wh_s, HH, H4, HH, H4, HH, /*2B*/1u);
        __builtin_amdgcn_s_wait_tensorcnt(0);
    }
    // init carry to zero while the DMA is in flight
    #pragma unroll
    for (int jj = 0; jj < 8; ++jj) {
        h_s[erow * HH + j0 + jj] = 0;
        c_s[erow * HH + j0 + jj] = 0.0f;
    }
    __syncthreads();

    for (int t = 0; t < TT; ++t) {
        // ---- mask reset of carry (before use) ----
        const float keep = 1.0f - (float)mask[(b0 + erow) * TT + t];
        #pragma unroll
        for (int jj = 0; jj < 8; ++jj) {
            int idx = erow * HH + j0 + jj;
            c_s[idx] *= keep;
            h_s[idx]  = f2bf(bf2f(h_s[idx]) * keep);
        }
        __syncthreads();

        // ---- z = u_t@Wx + h@Wh (each wave: 4 N-tiles of 16, K=128 each) ----
        const unsigned short* urow = u_bf + ((size_t)(b0 + ln) * TT + t) * HH;
        if (t + 1 < TT) __builtin_prefetch(urow + HH, 0, 1);   // next step's row
        v16bf au[4], ah[4];
        const unsigned short* hrow = h_s + ln * HH;
        #pragma unroll
        for (int kk = 0; kk < 4; ++kk) {
            au[kk] = load_a_bf16(urow, kk * 32, half);
            ah[kk] = load_a_bf16(hrow, kk * 32, half);
        }
        #pragma unroll
        for (int nt = 0; nt < 4; ++nt) {
            const int n0 = (wave * 4 + nt) * 16;
            const unsigned short* wx = wx_s + (size_t)(n0 + ln) * HH;
            const unsigned short* wh = wh_s + (size_t)(n0 + ln) * HH;
            v8f acc = {};
            #pragma unroll
            for (int kk = 0; kk < 4; ++kk)
                acc = wmma_bf16(au[kk], load_b_bf16(wx, kk * 32, half), acc);
            #pragma unroll
            for (int kk = 0; kk < 4; ++kk)
                acc = wmma_bf16(ah[kk], load_b_bf16(wh, kk * 32, half), acc);
            #pragma unroll
            for (int r = 0; r < 8; ++r)
                z_s[(r + 8 * half) * H4 + n0 + ln] = acc[r];
        }
        __syncthreads();

        // ---- gates ----
        #pragma unroll
        for (int jj = 0; jj < 8; ++jj) {
            const int j  = j0 + jj;
            const float iv = z_s[erow * H4 + j          ] + bc[j          ];
            const float fv = z_s[erow * H4 + j + HH     ] + bc[j + HH     ];
            const float gv = z_s[erow * H4 + j + 2 * HH ] + bc[j + 2 * HH ];
            const float ov = z_s[erow * H4 + j + 3 * HH ] + bc[j + 3 * HH ];
            const int   ci = erow * HH + j;
            const float c_new = sigm(fv) * c_s[ci] + sigm(iv) * tanh_fast(gv);
            const float h_new = sigm(ov) * tanh_fast(c_new);
            c_s[ci] = c_new;
            const unsigned short hb = f2bf(h_new);
            h_s[ci] = hb;
            r_bf[((size_t)(b0 + erow) * TT + t) * HH + j] = hb;
        }
        __syncthreads();
    }
}

// ---------------------------------------------------------------------------
// Kernel 3: out = relu(r@W2 + b2) @ W3 + b3.  4 waves/WG, one 16-row M tile
// per wave; y staged bf16 in LDS between the two GEMMs.
// ---------------------------------------------------------------------------
__global__ __launch_bounds__(128)
void k_head(const unsigned short* __restrict__ r_bf, const float* __restrict__ b2,
            const unsigned short* __restrict__ W2t, const float* __restrict__ b3,
            const unsigned short* __restrict__ W3t, float* __restrict__ out) {
    __shared__ __align__(16) unsigned short y_s[4][16 * HH];   // 16 KB

    const int tid  = threadIdx.x;
    const int wave = tid >> 5;
    const int lane = tid & 31;
    const int half = lane >> 4;
    const int ln   = lane & 15;
    const int wid  = blockIdx.x * 4 + wave;
    const int m0   = wid * 16;

    const unsigned short* rrow = r_bf + (size_t)(m0 + ln) * HH;
    v16bf afr[4];
    #pragma unroll
    for (int kk = 0; kk < 4; ++kk)
        afr[kk] = load_a_bf16(rrow, kk * 32, half);

    // y = relu(r@W2 + b2)  -> LDS bf16 (k contiguous per row)
    #pragma unroll
    for (int nt = 0; nt < 8; ++nt) {
        const int n0 = nt * 16;
        const unsigned short* wcol = W2t + (size_t)(n0 + ln) * HH;
        v8f acc = {};
        #pragma unroll
        for (int kk = 0; kk < 4; ++kk)
            acc = wmma_bf16(afr[kk], load_b_bf16(wcol, kk * 32, half), acc);
        const float bias = b2[n0 + ln];
        #pragma unroll
        for (int r = 0; r < 8; ++r)
            y_s[wave][(r + 8 * half) * HH + n0 + ln] = f2bf(fmaxf(acc[r] + bias, 0.0f));
    }
    __syncthreads();

    // out = y @ W3 + b3   (single N-tile, only columns 0..5 valid)
    const unsigned short* yrow = y_s[wave] + ln * HH;
    v8f acc = {};
    #pragma unroll
    for (int kk = 0; kk < 4; ++kk) {
        v16bf a = load_a_bf16(yrow, kk * 32, half);
        acc = wmma_bf16(a, load_b_bf16(W3t + (size_t)ln * HH, kk * 32, half), acc);
    }
    if (ln < AA) {
        const float bias = b3[ln];
        #pragma unroll
        for (int r = 0; r < 8; ++r)
            out[(size_t)(m0 + r + 8 * half) * AA + ln] = acc[r] + bias;
    }
}

// ---------------------------------------------------------------------------
// Launch
// ---------------------------------------------------------------------------
extern "C" void kernel_launch(void* const* d_in, const int* in_sizes, int n_in,
                              void* d_out, int out_size, void* d_ws, size_t ws_size,
                              hipStream_t stream) {
    const float* x    = (const float*)d_in[0];
    const float* W1   = (const float*)d_in[1];
    const float* b1   = (const float*)d_in[2];
    const float* Wx   = (const float*)d_in[3];
    const float* Wh   = (const float*)d_in[4];
    const float* bc   = (const float*)d_in[5];
    const float* W2   = (const float*)d_in[6];
    const float* b2   = (const float*)d_in[7];
    const float* W3   = (const float*)d_in[8];
    const float* b3   = (const float*)d_in[9];
    const int*   mask = (const int*)d_in[10];
    float*       out  = (float*)d_out;

    // workspace layout (all offsets 16B-aligned)
    char* w = (char*)d_ws;
    unsigned short* u_bf = (unsigned short*)w;  w += (size_t)BT * HH * 2;   // 32 MB
    unsigned short* r_bf = (unsigned short*)w;  w += (size_t)BT * HH * 2;   // 32 MB
    unsigned short* W1t  = (unsigned short*)w;  w += (size_t)HH * OBS * 2;  // [128][512]
    unsigned short* Wxt  = (unsigned short*)w;  w += (size_t)H4 * HH * 2;   // [512][128]
    unsigned short* Wht  = (unsigned short*)w;  w += (size_t)H4 * HH * 2;   // [512][128]
    unsigned short* W2t  = (unsigned short*)w;  w += (size_t)HH * HH * 2;   // [128][128]
    unsigned short* W3t  = (unsigned short*)w;  w += (size_t)16 * HH * 2;   // [16][128]

    // weight prep: transpose + bf16
    k_transpose_bf16<<<(HH * OBS + 255) / 256, 256, 0, stream>>>(W1, W1t, OBS, HH, HH);
    k_transpose_bf16<<<(H4 * HH + 255) / 256, 256, 0, stream>>>(Wx, Wxt, HH, H4, H4);
    k_transpose_bf16<<<(H4 * HH + 255) / 256, 256, 0, stream>>>(Wh, Wht, HH, H4, H4);
    k_transpose_bf16<<<(HH * HH + 255) / 256, 256, 0, stream>>>(W2, W2t, HH, HH, HH);
    k_transpose_bf16<<<(16 * HH + 255) / 256, 256, 0, stream>>>(W3, W3t, HH, AA, 16);

    // u = relu(x@W1 + b1): 8192 m-tiles, 8 waves/WG
    k_gemm1<<<BT / 16 / 8, 256, 0, stream>>>(x, b1, W1t, u_bf);

    // masked LSTM: 32 WGs of 8 waves, weights DMAed to LDS, T-loop inside
    k_lstm<<<BB / 16, 256, LSTM_LDS_BYTES, stream>>>(u_bf, Wxt, Wht, bc, mask, r_bf);

    // head: 4 waves/WG
    k_head<<<BT / 16 / 4, 128, 0, stream>>>(r_bf, b2, W2t, b3, W3t, out);
}
